// EncoderBlock_73770358276935
// MI455X (gfx1250) — matrix operand order repaired and checked by reading, not compiled, not run on hardware
//
#include <hip/hip_runtime.h>

// ---------------------------------------------------------------------------
// CDNA5 (gfx1250, wave32) windowed-attention transformer block.
// GEMMs on v_wmma_f32_16x16x32_bf16 (f32 accum); contiguous tile staging via
// global_load_async_to_lds_b128 (ASYNCcnt), conversions staged on VALU.
// ---------------------------------------------------------------------------

typedef __bf16 bf16;
typedef __attribute__((ext_vector_type(16))) __bf16 v16bf;
typedef __attribute__((ext_vector_type(8)))  float  v8f;

#define WSZ   14
#define NWIN  200            // 8 images * 5 * 5 windows
#define NTOK  196            // 14*14 tokens per window
#define NHEAD 12
#define HDIM  64
#define CDIM  768
#define MLPD  3072
#define TOKA  (NWIN * NTOK)  // 39200 window tokens
#define TOKB  (8 * 64 * 64)  // 32768 image tokens

// ---------------------------------------------------------------------------
// Async global->LDS copy (CDNA5, tracked by ASYNCcnt). LDS byte address is the
// low 32 bits of the generic pointer.
// ---------------------------------------------------------------------------
__device__ __forceinline__ unsigned lds_addr_of(const void* p) {
  return (unsigned)(size_t)p;
}
__device__ __forceinline__ void async_load_b128(unsigned lds_addr, const void* gaddr) {
  asm volatile("global_load_async_to_lds_b128 %0, %1, off"
               :: "v"(lds_addr), "v"(gaddr)
               : "memory");
}
__device__ __forceinline__ void wait_async0() {
  asm volatile("s_wait_asynccnt 0x0" ::: "memory");
}

// ---------------------------------------------------------------------------
// WMMA helpers: fragment layouts used by this code (wave32; A and B share one k order).
//  A (16x32 bf16): lane m = lane&15, half = lane>>4.
//    elems 0..7  -> k = half*16 + j       (contiguous 8 bf16)
//    elems 8..15 -> k = half*16 + 8 + j  (contiguous 8 bf16)
//  B (32x16 bf16), staged transposed in LDS as [n][k]:
//    lane n = lane&15, elems j -> k = (lane>>4)*16 + j (contiguous 16 bf16)
//  C/D (16x16 f32): row = (lane>>4)*8 + v, col = lane&15.
// ---------------------------------------------------------------------------
__device__ __forceinline__ v8f zero8() {
  v8f z;
#pragma unroll
  for (int i = 0; i < 8; ++i) z[i] = 0.0f;
  return z;
}

__device__ __forceinline__ v16bf frag_a_lds(const bf16* base, int lda, int lane) {
  const int m = lane & 15, h = lane >> 4;
  const bf16* r0 = base + m * lda + h * 16;
  const bf16* r1 = base + m * lda + h * 16 + 8;
  v16bf f;
#pragma unroll
  for (int j = 0; j < 8; ++j) f[j] = r0[j];
#pragma unroll
  for (int j = 0; j < 8; ++j) f[8 + j] = r1[j];
  return f;
}

__device__ __forceinline__ v16bf frag_b_lds(const bf16* baseT, int ldk, int lane) {
  const int n = lane & 15, h = lane >> 4;
  const bf16* r = baseT + n * ldk + h * 16;
  v16bf f;
#pragma unroll
  for (int j = 0; j < 16; ++j) f[j] = r[j];
  return f;
}

__device__ __forceinline__ v8f wmma_bf16(v16bf a, v16bf b, v8f c) {
  return __builtin_amdgcn_wmma_f32_16x16x32_bf16(
      /*neg_a=*/false, a, /*neg_b=*/false, b,
      /*c_mod=*/(short)0, c, /*reuse_a=*/false, /*reuse_b=*/false);
}

// ---------------------------------------------------------------------------
// Kernel 1: LN1 + window partition -> bf16 [TOKA, CDIM]. One wave per token.
// ---------------------------------------------------------------------------
__global__ __launch_bounds__(256)
void ln1_window_kernel(const float* __restrict__ x, const float* __restrict__ g,
                       const float* __restrict__ b, bf16* __restrict__ xwin) {
  const int wv = threadIdx.x >> 5, lane = threadIdx.x & 31;
  const int gt = blockIdx.x * 8 + wv;
  if (gt >= TOKA) return;
  const int w = gt / NTOK, n = gt % NTOK;
  const int bb = w / 25, rem = w % 25, wh = rem / 5, ww = rem % 5;
  const int ih = n / WSZ, iw = n % WSZ;
  const int gh = wh * WSZ + ih, gw = ww * WSZ + iw;
  bf16* dst = xwin + (size_t)gt * CDIM;
  if (gh >= 64 || gw >= 64) {
#pragma unroll
    for (int i = 0; i < 24; ++i) dst[lane + i * 32] = (bf16)0.0f;
    return;
  }
  const float* src = x + (size_t)((bb * 64 + gh) * 64 + gw) * CDIM;
  float vals[24];
  float s = 0.0f;
#pragma unroll
  for (int i = 0; i < 24; ++i) { float v = src[lane + i * 32]; vals[i] = v; s += v; }
#pragma unroll
  for (int o = 16; o > 0; o >>= 1) s += __shfl_xor(s, o, 32);
  const float mu = s * (1.0f / 768.0f);
  float q = 0.0f;
#pragma unroll
  for (int i = 0; i < 24; ++i) { float d = vals[i] - mu; q += d * d; }
#pragma unroll
  for (int o = 16; o > 0; o >>= 1) q += __shfl_xor(q, o, 32);
  const float rstd = rsqrtf(q * (1.0f / 768.0f) + 1e-6f);
#pragma unroll
  for (int i = 0; i < 24; ++i) {
    const int c = lane + i * 32;
    dst[c] = (bf16)((vals[i] - mu) * rstd * g[c] + b[c]);
  }
}

// ---------------------------------------------------------------------------
// Kernel 2: generic tiled bf16 WMMA GEMM, C = A[M,K] (bf16) * B[K,N] (f32,
// converted on LDS stage) + bias, with templated epilogue.
// 256 threads = 8 waves; macro-tile 128x128; each wave owns 64x32 (4x2 tiles).
// A tiles staged with async global->LDS B128; B tiles converted on VALU.
// ---------------------------------------------------------------------------
enum { EPI_QKV = 0, EPI_GELU = 1, EPI_F32 = 2, EPI_F32RES = 3, EPI_BF16 = 4 };

template <int EPI>
__global__ __launch_bounds__(256)
void gemm_kernel(const bf16* __restrict__ A, const float* __restrict__ B,
                 const float* __restrict__ bias, const float* __restrict__ res,
                 void* __restrict__ outv, int M, int N, int K) {
  __shared__ bf16 As[128 * 32];  // [row][k], row-major (64 B per row)
  __shared__ bf16 Bt[128 * 32];  // [n][k], transposed for B fragments
  const int t = threadIdx.x, lane = t & 31, wv = t >> 5;
  const int m0 = blockIdx.y * 128, n0 = blockIdx.x * 128;
  const int wm = (wv >> 2) * 64, wn = (wv & 3) * 32;
  const unsigned as_base = lds_addr_of(As);

  v8f acc[4][2];
#pragma unroll
  for (int i = 0; i < 4; ++i)
#pragma unroll
    for (int j = 0; j < 2; ++j) acc[i][j] = zero8();

  for (int k0 = 0; k0 < K; k0 += 32) {
    // --- stage A tile: 512 x 16B chunks, async copy (rows are contiguous) ---
#pragma unroll
    for (int u = 0; u < 2; ++u) {
      const int cc = t + u * 256;          // chunk id 0..511
      const int row = cc >> 2, sub = (cc & 3) * 16;  // byte offset in row
      const int gm = m0 + row;
      if (gm < M) {
        async_load_b128(as_base + row * 64 + sub,
                        (const char*)A + ((size_t)gm * K + k0) * 2 + sub);
      } else {
        bf16* z = As + row * 32 + (sub >> 1);
#pragma unroll
        for (int j = 0; j < 8; ++j) z[j] = (bf16)0.0f;
      }
    }
    // --- stage B tile transposed, f32 -> bf16 on VALU ---
#pragma unroll
    for (int i = 0; i < 16; ++i) {
      const int e = t + i * 256;
      const int k = e >> 7, n = e & 127;
      const int gn = n0 + n;
      const float v = (gn < N) ? B[(size_t)(k0 + k) * N + gn] : 0.0f;
      Bt[n * 32 + k] = (bf16)v;
    }
    wait_async0();
    __syncthreads();
    v16bf af[4], bfr[2];
#pragma unroll
    for (int i = 0; i < 4; ++i) af[i] = frag_a_lds(As + (wm + i * 16) * 32, 32, lane);
#pragma unroll
    for (int j = 0; j < 2; ++j) bfr[j] = frag_b_lds(Bt + (wn + j * 16) * 32, 32, lane);
#pragma unroll
    for (int i = 0; i < 4; ++i)
#pragma unroll
      for (int j = 0; j < 2; ++j) acc[i][j] = wmma_bf16(af[i], bfr[j], acc[i][j]);
    __syncthreads();
  }

  const int half = lane >> 4, l16 = lane & 15;
#pragma unroll
  for (int j = 0; j < 2; ++j) {
    const int gn = n0 + wn + j * 16 + l16;
    if (gn >= N) continue;
    const float bv = bias[gn];
    // hoisted column decomposition for the QKV scatter
    const int which = gn / CDIM, remc = gn % CDIM;
    const int hh = remc / HDIM, dd = remc % HDIM;
#pragma unroll
    for (int i = 0; i < 4; ++i)
#pragma unroll
      for (int v = 0; v < 8; ++v) {
        const int gm = m0 + wm + i * 16 + half * 8 + v;
        if (gm >= M) continue;
        const float val = acc[i][j][v] + bv;
        if constexpr (EPI == EPI_QKV) {
          bf16* out = (bf16*)outv;
          const int w = gm / NTOK, nn = gm % NTOK;
          const size_t dst = (size_t)which * ((size_t)NWIN * NHEAD * NTOK * HDIM) +
                             ((size_t)((w * NHEAD + hh) * NTOK + nn)) * HDIM + dd;
          out[dst] = (bf16)val;
        } else if constexpr (EPI == EPI_GELU) {
          bf16* out = (bf16*)outv;
          const float ge = 0.5f * val * (1.0f + erff(val * 0.70710678118654752f));
          out[(size_t)gm * N + gn] = (bf16)ge;
        } else if constexpr (EPI == EPI_F32) {
          ((float*)outv)[(size_t)gm * N + gn] = val;
        } else if constexpr (EPI == EPI_F32RES) {
          ((float*)outv)[(size_t)gm * N + gn] = val + res[(size_t)gm * N + gn];
        } else {
          ((bf16*)outv)[(size_t)gm * N + gn] = (bf16)val;
        }
      }
  }
}

// ---------------------------------------------------------------------------
// Kernel 3: attention, one workgroup per (window, head). K async-copied to
// LDS, V transposed on VALU; 16-row query tiles: S = Q K^T (WMMA) + decomposed
// rel-pos bias, softmax, O = P V (WMMA). Keys padded to 224 with zeros.
// Dynamic LDS ~83 KB.
// ---------------------------------------------------------------------------
__global__ __launch_bounds__(256)
void attn_kernel(const bf16* __restrict__ Qg, const bf16* __restrict__ Kg,
                 const bf16* __restrict__ Vg, const float* __restrict__ rph,
                 const float* __restrict__ rpw, bf16* __restrict__ out) {
  extern __shared__ char smem[];
  bf16* Ks = (bf16*)smem;        // [224][64]  key tokens, row-major
  bf16* Vt = Ks + 224 * 64;      // [64][224]  V transposed (B-operand layout)
  bf16* Qs = Vt + 64 * 224;      // [16][64]   query tile
  bf16* Pb = Qs + 16 * 64;       // [16][224]  softmax probs (bf16)
  float* Sb = (float*)(Pb + 16 * 224);  // [16][224] logits
  float* rH = Sb + 16 * 224;     // [16][14]
  float* rW = rH + 224;          // [16][14]

  const int t = threadIdx.x, lane = t & 31, wv = t >> 5;
  const int wid = blockIdx.x;               // window*NHEAD + head
  const int w = wid / NHEAD, h = wid % NHEAD;
  const size_t base = (size_t)wid * NTOK * HDIM;
  const unsigned ks_base = lds_addr_of(Ks);
  const unsigned qs_base = lds_addr_of(Qs);

  // K rows 0..195: one contiguous 25088-byte async copy (1568 x 16B chunks)
  for (int cc = t; cc < 1568; cc += 256)
    async_load_b128(ks_base + cc * 16, (const char*)(Kg + base) + cc * 16);
  // zero pad rows 196..223
  for (int e = t; e < 28 * 64; e += 256) Ks[196 * 64 + e] = (bf16)0.0f;
  // V transposed staging on VALU
#pragma unroll
  for (int i = 0; i < 56; ++i) {
    const int e = t + i * 256;
    const int d = e / 224, tok = e % 224;
    Vt[e] = (tok < NTOK) ? Vg[base + tok * HDIM + d] : (bf16)0.0f;
  }
  wait_async0();
  __syncthreads();

  for (int t0 = 0; t0 < NTOK; t0 += 16) {
    // stage Q tile: async copy of valid bytes, zero-fill the tail
    const int vrows = (NTOK - t0 < 16) ? (NTOK - t0) : 16;
    const int vbytes = vrows * 128;
    if (t < 128) {
      const int off = t * 16;
      if (off < vbytes)
        async_load_b128(qs_base + off,
                        (const char*)(Qg + base) + (size_t)t0 * 128 + off);
    }
    for (int e = t; e < 1024; e += 256)
      if ((e >> 6) >= vrows) Qs[e] = (bf16)0.0f;
    wait_async0();
    __syncthreads();

    // decomposed rel-pos bias: relH[r][a] = q_r . rel_pos_h[ih - a + 13]
    for (int task = t; task < 448; task += 256) {
      const int sel = task / 224, rr = (task % 224) / WSZ, a = task % WSZ;
      const int i = t0 + rr;
      float accd = 0.0f;
      if (i < NTOK) {
        const int ih = i / WSZ, iw = i % WSZ;
        const int row = (sel ? iw : ih) - a + (WSZ - 1);
        const float* R = (sel ? rpw : rph) + row * HDIM;
        for (int c = 0; c < HDIM; ++c) accd += (float)Qs[rr * HDIM + c] * R[c];
      }
      (sel ? rW : rH)[rr * WSZ + a] = accd;
    }
    __syncthreads();

    // S = Q K^T over 14 column tiles (224 padded keys)
    for (int nt = wv; nt < 14; nt += 8) {
      v8f accs = zero8();
#pragma unroll
      for (int c = 0; c < 2; ++c) {
        v16bf a = frag_a_lds(Qs + c * 32, HDIM, lane);
        v16bf b = frag_b_lds(Ks + nt * 16 * HDIM + c * 32, HDIM, lane);
        accs = wmma_bf16(a, b, accs);
      }
      const int half = lane >> 4, l16 = lane & 15;
#pragma unroll
      for (int v = 0; v < 8; ++v)
        Sb[(half * 8 + v) * 224 + nt * 16 + l16] = accs[v];
    }
    __syncthreads();

    // softmax with scale 1/8 + rel bias; 16 threads per row
    {
      const int r = t >> 4, l16 = t & 15;
      float mx = -3.0e38f;
      for (int j = l16; j < NTOK; j += 16) {
        const float val = Sb[r * 224 + j] * 0.125f +
                          rH[r * WSZ + j / WSZ] + rW[r * WSZ + j % WSZ];
        Sb[r * 224 + j] = val;
        mx = fmaxf(mx, val);
      }
#pragma unroll
      for (int o = 8; o > 0; o >>= 1) mx = fmaxf(mx, __shfl_xor(mx, o, 16));
      float s = 0.0f;
      for (int j = l16; j < NTOK; j += 16) {
        const float e = __expf(Sb[r * 224 + j] - mx);
        Sb[r * 224 + j] = e;
        s += e;
      }
#pragma unroll
      for (int o = 8; o > 0; o >>= 1) s += __shfl_xor(s, o, 16);
      const float inv = 1.0f / s;
      for (int j = l16; j < 224; j += 16)
        Pb[r * 224 + j] = (j < NTOK) ? (bf16)(Sb[r * 224 + j] * inv) : (bf16)0.0f;
    }
    __syncthreads();

    // O = P V : 4 column tiles of HDIM, waves 0..3
    if (wv < 4) {
      v8f acco = zero8();
#pragma unroll
      for (int ch = 0; ch < 7; ++ch) {
        v16bf a = frag_a_lds(Pb + ch * 32, 224, lane);
        v16bf b = frag_b_lds(Vt + wv * 16 * 224 + ch * 32, 224, lane);
        acco = wmma_bf16(a, b, acco);
      }
      const int half = lane >> 4, l16 = lane & 15;
#pragma unroll
      for (int v = 0; v < 8; ++v) {
        const int tok = t0 + half * 8 + v;
        if (tok < NTOK)
          out[((size_t)(w * NTOK + tok)) * CDIM + h * HDIM + wv * 16 + l16] =
              (bf16)acco[v];
      }
    }
    __syncthreads();
  }
}

// ---------------------------------------------------------------------------
// Kernel 4: window unpartition + residual + LN2. Writes x1 (f32) and LN2
// output (bf16 MLP input). One wave per image token.
// ---------------------------------------------------------------------------
__global__ __launch_bounds__(256)
void resid_ln2_kernel(const float* __restrict__ x, const float* __restrict__ proj,
                      const float* __restrict__ g, const float* __restrict__ b,
                      float* __restrict__ x1, bf16* __restrict__ y) {
  const int wv = threadIdx.x >> 5, lane = threadIdx.x & 31;
  const int gt = blockIdx.x * 8 + wv;
  if (gt >= TOKB) return;
  const int bb = gt / 4096, rem = gt % 4096, gh = rem / 64, gw = rem % 64;
  const int wh = gh / WSZ, ih = gh % WSZ, ww = gw / WSZ, iw = gw % WSZ;
  const int w = (bb * 5 + wh) * 5 + ww, n = ih * WSZ + iw;
  const float* px = x + (size_t)gt * CDIM;
  const float* pp = proj + ((size_t)(w * NTOK + n)) * CDIM;
  float* px1 = x1 + (size_t)gt * CDIM;
  bf16* py = y + (size_t)gt * CDIM;
  float vals[24];
  float s = 0.0f;
#pragma unroll
  for (int i = 0; i < 24; ++i) {
    const int c = lane + i * 32;
    const float v = px[c] + pp[c];
    px1[c] = v;
    vals[i] = v;
    s += v;
  }
#pragma unroll
  for (int o = 16; o > 0; o >>= 1) s += __shfl_xor(s, o, 32);
  const float mu = s * (1.0f / 768.0f);
  float q = 0.0f;
#pragma unroll
  for (int i = 0; i < 24; ++i) { float d = vals[i] - mu; q += d * d; }
#pragma unroll
  for (int o = 16; o > 0; o >>= 1) q += __shfl_xor(q, o, 32);
  const float rstd = rsqrtf(q * (1.0f / 768.0f) + 1e-6f);
#pragma unroll
  for (int i = 0; i < 24; ++i) {
    const int c = lane + i * 32;
    py[c] = (bf16)((vals[i] - mu) * rstd * g[c] + b[c]);
  }
}

// ---------------------------------------------------------------------------
extern "C" void kernel_launch(void* const* d_in, const int* in_sizes, int n_in,
                              void* d_out, int out_size, void* d_ws, size_t ws_size,
                              hipStream_t stream) {
  const float* x      = (const float*)d_in[0];
  const float* ln1_g  = (const float*)d_in[1];
  const float* ln1_b  = (const float*)d_in[2];
  const float* qkv_w  = (const float*)d_in[3];
  const float* qkv_b  = (const float*)d_in[4];
  const float* proj_w = (const float*)d_in[5];
  const float* proj_b = (const float*)d_in[6];
  const float* rph    = (const float*)d_in[7];
  const float* rpw    = (const float*)d_in[8];
  const float* ln2_g  = (const float*)d_in[9];
  const float* ln2_b  = (const float*)d_in[10];
  const float* mlp_w1 = (const float*)d_in[11];
  const float* mlp_b1 = (const float*)d_in[12];
  const float* mlp_w2 = (const float*)d_in[13];
  const float* mlp_b2 = (const float*)d_in[14];

  char* ws = (char*)d_ws;
  const size_t SZ_XWIN = (size_t)TOKA * CDIM * sizeof(bf16);                  // 60.2 MB
  const size_t SZ_Q1   = (size_t)NWIN * NHEAD * NTOK * HDIM * sizeof(bf16);   // 60.2 MB
  const size_t SZ_PROJ = (size_t)TOKA * CDIM * sizeof(float);                 // 120.4 MB
  const size_t SZ_X1   = (size_t)TOKB * CDIM * sizeof(float);                 // 100.7 MB

  bf16*  xwin     = (bf16*)ws;                                  // dead after QKV gemm
  bf16*  qkv      = (bf16*)(ws + SZ_XWIN);                      // Q,K,V contiguous
  bf16*  attn_out = xwin;                                       // reuses xwin region
  float* proj_out = (float*)(ws + SZ_XWIN + 3 * SZ_Q1);
  float* x1       = (float*)((char*)proj_out + SZ_PROJ);
  bf16*  ybuf     = (bf16*)((char*)x1 + SZ_X1);
  bf16*  m1       = (bf16*)ws;  // mlp1 out (201 MB) reuses dead xwin+qkv (240 MB)

  // 1. LN1 + window partition
  ln1_window_kernel<<<(TOKA + 7) / 8, 256, 0, stream>>>(x, ln1_g, ln1_b, xwin);

  // 2. QKV GEMM: [39200,768] x [768,2304]
  gemm_kernel<EPI_QKV><<<dim3((3 * CDIM) / 128, (TOKA + 127) / 128), 256, 0, stream>>>(
      xwin, qkv_w, qkv_b, nullptr, (void*)qkv, TOKA, 3 * CDIM, CDIM);

  // 3. windowed attention with decomposed rel-pos bias
  const bf16* Qg = qkv;
  const bf16* Kg = qkv + (size_t)NWIN * NHEAD * NTOK * HDIM;
  const bf16* Vg = Kg + (size_t)NWIN * NHEAD * NTOK * HDIM;
  const int smem_attn = (224 * 64 + 64 * 224 + 16 * 64 + 16 * 224) * 2 +
                        (16 * 224 + 224 + 224) * 4;  // 82688 B
  hipFuncSetAttribute((const void*)attn_kernel,
                      hipFuncAttributeMaxDynamicSharedMemorySize, smem_attn);
  attn_kernel<<<NWIN * NHEAD, 256, smem_attn, stream>>>(Qg, Kg, Vg, rph, rpw, attn_out);

  // 4. output projection: [39200,768] x [768,768] -> f32
  gemm_kernel<EPI_F32><<<dim3(CDIM / 128, (TOKA + 127) / 128), 256, 0, stream>>>(
      attn_out, proj_w, proj_b, nullptr, (void*)proj_out, TOKA, CDIM, CDIM);

  // 5. unpartition + residual + LN2
  resid_ln2_kernel<<<TOKB / 8, 256, 0, stream>>>(x, proj_out, ln2_g, ln2_b, x1, ybuf);

  // 6. MLP fc1 + exact GELU: [32768,768] x [768,3072]
  gemm_kernel<EPI_GELU><<<dim3(MLPD / 128, TOKB / 128), 256, 0, stream>>>(
      ybuf, mlp_w1, mlp_b1, nullptr, (void*)m1, TOKB, MLPD, CDIM);

  // 7. MLP fc2 + residual add -> d_out (f32): [32768,3072] x [3072,768]
  gemm_kernel<EPI_F32RES><<<dim3(CDIM / 128, TOKB / 128), 256, 0, stream>>>(
      m1, mlp_w2, mlp_b2, x1, d_out, TOKB, CDIM, MLPD);
}
